// MultiAttention_30348238913611
// MI455X (gfx1250) — matrix operand order repaired and checked by reading, not compiled, not run on hardware
//
#include <hip/hip_runtime.h>

// ---------------------------------------------------------------------------
// Problem dims (from reference): L=2048, N=4, E=1024, H=8
// ---------------------------------------------------------------------------
static constexpr int Ld  = 2048;
static constexpr int Nd  = 4;
static constexpr int Ed  = 1024;
static constexpr int Hd  = 8;
static constexpr int Td  = Ld * Nd;       // 8192 tokens
static constexpr int F3  = 3 * Ed;        // 3072
static constexpr int BHN = Hd * Nd;       // 32 (h,n) pairs

// Workspace layout (bytes)
static constexpr size_t OFF_XB    = 0;                                    // [Td][Ed] bf16
static constexpr size_t OFF_WQKVB = OFF_XB    + (size_t)Td * Ed * 2;      // [Hd][F3][Ed] bf16
static constexpr size_t OFF_WOB   = OFF_WQKVB + (size_t)Hd * F3 * Ed * 2; // [Hd][Ed][Ed] bf16
static constexpr size_t OFF_QKV   = OFF_WOB   + (size_t)Hd * Ed * Ed * 2; // [Hd][Td][F3] bf16
static constexpr size_t OFF_SC    = OFF_QKV   + (size_t)Hd * Td * F3 * 2; // [BHN][Ld][Ld] f32
static constexpr size_t OFF_AT    = OFF_SC    + (size_t)BHN * Ld * Ld * 4;// [BHN][Ld][Ld] bf16
static constexpr size_t OFF_VT    = OFF_AT    + (size_t)BHN * Ld * Ld * 2;// [BHN][Ed][Ld] bf16
static constexpr size_t OFF_CTX   = OFF_VT    + (size_t)BHN * Ed * Ld * 2;// [Hd][Td][Ed] bf16

typedef __attribute__((ext_vector_type(16))) __bf16 bf16x16;
typedef __attribute__((ext_vector_type(8)))  __bf16 bf16x8;
typedef __attribute__((ext_vector_type(4)))  __bf16 bf16x4;
typedef __attribute__((ext_vector_type(8)))  float  f32x8;

#define DEV static __device__ __forceinline__

DEV __bf16 f2bf(float f) {
    unsigned u = __builtin_bit_cast(unsigned, f);
    u += 0x7FFFu + ((u >> 16) & 1u);          // round-to-nearest-even
    unsigned short s = (unsigned short)(u >> 16);
    return __builtin_bit_cast(__bf16, s);
}

// A fragment (16x32 bf16, row-major source, row stride lda elements).
// ISA 7.12.2: lane<16 holds row M=lane, K={0..7} U {16..23}; lane>=16: K={8..15} U {24..31}.
DEV bf16x16 load_fragA(const __bf16* base, long lda, int row, int k) {
    const int lane = threadIdx.x & 31;
    const __bf16* p = base + (long)(row + (lane & 15)) * lda + k + ((lane >> 4) << 3);
    bf16x8 lo = *reinterpret_cast<const bf16x8*>(p);
    bf16x8 hi = *reinterpret_cast<const bf16x8*>(p + 16);
    bf16x16 r;
#pragma unroll
    for (int i = 0; i < 8; ++i) { r[i] = lo[i]; r[i + 8] = hi[i]; }
    return r;
}

// B fragment (32x16 bf16) loaded from Bt = B-transposed rows (row = output column,
// K contiguous along the row). lane<16: K={0..7} U {16..23}; lane>=16: K={8..15} U {24..31}.
DEV bf16x16 load_fragB(const __bf16* bt, long ldb, int col, int k) {
    const int lane = threadIdx.x & 31;
    const __bf16* p = bt + (long)(col + (lane & 15)) * ldb + k + ((lane >> 4) << 3);
    bf16x8 lo = *reinterpret_cast<const bf16x8*>(p);
    bf16x8 hi = *reinterpret_cast<const bf16x8*>(p + 16);
    bf16x16 r;
#pragma unroll
    for (int i = 0; i < 8; ++i) { r[i] = lo[i]; r[i + 8] = hi[i]; }
    return r;
}

DEV f32x8 wmma_bf16(bf16x16 a, bf16x16 b, f32x8 c) {
    return __builtin_amdgcn_wmma_f32_16x16x32_bf16(false, a, false, b, (short)0, c,
                                                   false, false);
}

// LDS byte offset of a generic pointer to __shared__ (low 32 bits of the
// aperture-based generic address are the LDS offset).
DEV unsigned lds_off(const void* p) { return (unsigned)(unsigned long long)p; }

// Async-copy one K-step (32 halves) of tiles into LDS:
//   A tile: 256 rows x 32 halves (16 KB) -> 4 rounds of 256 lanes x 16B
//   B tile:  64 rows x 32 halves ( 4 KB) -> 1 round  of 256 lanes x 16B
// Uses GLOBAL_LOAD_ASYNC_TO_LDS_B128 (ASYNCcnt), per-lane 16-byte copies.
DEV void stage_async(const __bf16* gA, long lda, int row0,
                     const __bf16* gBt, long ldb, int col0, int k,
                     __bf16* sA, __bf16* sB) {
    const int tid = threadIdx.x;
#pragma unroll
    for (int r = 0; r < 4; ++r) {
        const int off = (tid + r * 256) * 8;      // halves within 256x32 tile
        const int row = off >> 5, col = off & 31;
        const unsigned lo = lds_off(sA + row * 32 + col);
        const __bf16* g = gA + (long)(row0 + row) * lda + k + col;
        asm volatile("global_load_async_to_lds_b128 %0, %1, off"
                     :: "v"(lo), "v"(g) : "memory");
    }
    {
        const int off = tid * 8;                  // halves within 64x32 tile
        const int row = off >> 5, col = off & 31;
        const unsigned lo = lds_off(sB + row * 32 + col);
        const __bf16* g = gBt + (long)(col0 + row) * ldb + k + col;
        asm volatile("global_load_async_to_lds_b128 %0, %1, off"
                     :: "v"(lo), "v"(g) : "memory");
    }
}

DEV void wait_async0() { asm volatile("s_wait_asynccnt 0" ::: "memory"); }

// Block-level GEMM: 256 (rows) x 64 (cols) output tile per 256-thread block,
// double-buffered LDS staging (40 KB), K-step 32. Each wave computes a 32x64
// slice: 2 A fragments sharing 4 B fragments -> 8 WMMAs per 12 ds_load_b128.
// acc[][] persists across calls (used for the head-summed output projection).
DEV void block_gemm_lds(const __bf16* A, long lda, const __bf16* Bt, long ldb,
                        int row0, int col0, int K,
                        __bf16 (*sA)[256 * 32], __bf16 (*sB)[64 * 32],
                        f32x8 acc[2][4]) {
    const int wave = threadIdx.x >> 5;
    stage_async(A, lda, row0, Bt, ldb, col0, 0, sA[0], sB[0]);
    wait_async0();
    __syncthreads();
    const int nk = K >> 5;
    for (int kt = 0; kt < nk; ++kt) {
        const int cur = kt & 1;
        if (kt + 1 < nk)   // prefetch next K-step while computing this one
            stage_async(A, lda, row0, Bt, ldb, col0, (kt + 1) << 5,
                        sA[cur ^ 1], sB[cur ^ 1]);
        // Load all fragments first, then issue 8 back-to-back WMMAs.
        bf16x16 a0 = load_fragA(sA[cur], 32, wave * 32, 0);
        bf16x16 a1 = load_fragA(sA[cur], 32, wave * 32 + 16, 0);
        bf16x16 b0 = load_fragB(sB[cur], 32, 0, 0);
        bf16x16 b1 = load_fragB(sB[cur], 32, 16, 0);
        bf16x16 b2 = load_fragB(sB[cur], 32, 32, 0);
        bf16x16 b3 = load_fragB(sB[cur], 32, 48, 0);
        acc[0][0] = wmma_bf16(a0, b0, acc[0][0]);
        acc[0][1] = wmma_bf16(a0, b1, acc[0][1]);
        acc[0][2] = wmma_bf16(a0, b2, acc[0][2]);
        acc[0][3] = wmma_bf16(a0, b3, acc[0][3]);
        acc[1][0] = wmma_bf16(a1, b0, acc[1][0]);
        acc[1][1] = wmma_bf16(a1, b1, acc[1][1]);
        acc[1][2] = wmma_bf16(a1, b2, acc[1][2]);
        acc[1][3] = wmma_bf16(a1, b3, acc[1][3]);
        if (kt + 1 < nk) wait_async0();
        __syncthreads();
    }
}

// ---------------------------------------------------------------------------
// 1) f32 -> bf16 conversion
// ---------------------------------------------------------------------------
__global__ __launch_bounds__(256) void k_cvt_bf16(const float* __restrict__ in,
                                                  __bf16* __restrict__ out, long n) {
    long i = ((long)blockIdx.x * 256 + threadIdx.x) * 4;
    if (i >= n) return;
    float4 f = *reinterpret_cast<const float4*>(in + i);
    bf16x4 o;
    o[0] = f2bf(f.x); o[1] = f2bf(f.y); o[2] = f2bf(f.z); o[3] = f2bf(f.w);
    *reinterpret_cast<bf16x4*>(out + i) = o;
}

// ---------------------------------------------------------------------------
// 2) QKV projection: per head h, QKV[h] = Xb * Wqkv[h]^T + bqkv[h]  (bf16 out)
//    M=Td(8192), N=F3(3072), K=Ed(1024)
// ---------------------------------------------------------------------------
__global__ __launch_bounds__(256) void k_gemm_qkv(const __bf16* __restrict__ X,
                                                  const __bf16* __restrict__ W,
                                                  const float* __restrict__ bqkv,
                                                  __bf16* __restrict__ QKV) {
    __shared__ alignas(16) __bf16 sA[2][256 * 32];
    __shared__ alignas(16) __bf16 sB[2][64 * 32];
    const int h    = blockIdx.z;
    const int wave = threadIdx.x >> 5;
    const int lane = threadIdx.x & 31;
    const int row0 = blockIdx.y * 256;
    const int col0 = blockIdx.x * 64;

    const __bf16* Bt = W + (long)h * F3 * Ed;       // Wqkv[h] rows = output cols
    __bf16* C = QKV + (long)h * Td * F3;

    f32x8 acc[2][4] = {};
    block_gemm_lds(X, Ed, Bt, Ed, row0, col0, Ed, sA, sB, acc);

    const int cn = lane & 15, rh = (lane >> 4) * 8;
#pragma unroll
    for (int rr = 0; rr < 2; ++rr) {
        const int row = row0 + wave * 32 + rr * 16;
#pragma unroll
        for (int j = 0; j < 4; ++j) {
            const int col = col0 + 16 * j + cn;
            const float bias = bqkv[h * F3 + col];
#pragma unroll
            for (int i = 0; i < 8; ++i)
                C[(long)(row + rh + i) * F3 + col] = f2bf(acc[rr][j][i] + bias);
        }
    }
}

// ---------------------------------------------------------------------------
// 3) Scores: per (h,n), S[l,m] = q[l,:] . k[m,:]   (f32 out; scale applied in softmax)
//    q row l at QKV[h][(l*Nd+n)][0:Ed], k row m at QKV[h][(m*Nd+n)][Ed:2Ed]
// ---------------------------------------------------------------------------
__global__ __launch_bounds__(256) void k_gemm_scores(const __bf16* __restrict__ QKV,
                                                     float* __restrict__ S) {
    __shared__ alignas(16) __bf16 sA[2][256 * 32];
    __shared__ alignas(16) __bf16 sB[2][64 * 32];
    const int b = blockIdx.z, h = b >> 2, n = b & 3;
    const int wave = threadIdx.x >> 5;
    const int lane = threadIdx.x & 31;
    const int row0 = blockIdx.y * 256;   // l
    const int col0 = blockIdx.x * 64;    // m

    const __bf16* A  = QKV + (long)h * Td * F3 + (long)n * F3;        // q
    const __bf16* Bt = A + Ed;                                         // k
    float* C = S + (long)b * Ld * Ld;

    f32x8 acc[2][4] = {};
    block_gemm_lds(A, (long)Nd * F3, Bt, (long)Nd * F3, row0, col0, Ed, sA, sB, acc);

    const int cn = lane & 15, rh = (lane >> 4) * 8;
#pragma unroll
    for (int rr = 0; rr < 2; ++rr) {
        const int row = row0 + wave * 32 + rr * 16;
#pragma unroll
        for (int j = 0; j < 4; ++j)
#pragma unroll
            for (int i = 0; i < 8; ++i)
                C[(long)(row + rh + i) * Ld + col0 + 16 * j + cn] = acc[rr][j][i];
    }
}

// ---------------------------------------------------------------------------
// 4) Row softmax over 2048 scores (f32) -> bf16 attn, scale = 1/sqrt(E) folded in
// ---------------------------------------------------------------------------
__global__ __launch_bounds__(256) void k_softmax(const float* __restrict__ S,
                                                 __bf16* __restrict__ P) {
    __shared__ float red[256];
    const long row = blockIdx.x;
    const int tid = threadIdx.x;
    const float* s = S + row * Ld;
    __bf16* p = P + row * Ld;
    const float scale = 0.03125f;   // 1/32 = 1/sqrt(1024)

    float v[8];
    float mx = -3.0e38f;
#pragma unroll
    for (int j = 0; j < 8; ++j) { v[j] = s[tid + j * 256] * scale; mx = fmaxf(mx, v[j]); }
    red[tid] = mx; __syncthreads();
    for (int off = 128; off > 0; off >>= 1) {
        if (tid < off) red[tid] = fmaxf(red[tid], red[tid + off]);
        __syncthreads();
    }
    mx = red[0]; __syncthreads();

    float sum = 0.f;
#pragma unroll
    for (int j = 0; j < 8; ++j) { v[j] = __expf(v[j] - mx); sum += v[j]; }
    red[tid] = sum; __syncthreads();
    for (int off = 128; off > 0; off >>= 1) {
        if (tid < off) red[tid] += red[tid + off];
        __syncthreads();
    }
    const float inv = 1.0f / red[0];
#pragma unroll
    for (int j = 0; j < 8; ++j) p[tid + j * 256] = f2bf(v[j] * inv);
}

// ---------------------------------------------------------------------------
// 5) Transpose V per (h,n): vt[b][e][l] = QKV[h][(l*Nd+n)][2E + e]  (LDS tiled)
// ---------------------------------------------------------------------------
__global__ __launch_bounds__(256) void k_transpose_v(const __bf16* __restrict__ QKV,
                                                     __bf16* __restrict__ VT) {
    __shared__ __bf16 tile[32][33];
    const int b = blockIdx.z, h = b >> 2, n = b & 3;
    const int l0 = blockIdx.x * 32;   // key/seq dim
    const int e0 = blockIdx.y * 32;   // feature dim
    const __bf16* src = QKV + (long)h * Td * F3 + (long)n * F3 + 2 * Ed;
    __bf16* dst = VT + (long)b * Ed * Ld;

#pragma unroll
    for (int j = 0; j < 4; ++j) {
        const int l = l0 + threadIdx.y * 4 + j;
        tile[threadIdx.y * 4 + j][threadIdx.x] = src[(long)l * (Nd * F3) + e0 + threadIdx.x];
    }
    __syncthreads();
#pragma unroll
    for (int j = 0; j < 4; ++j) {
        const int e = e0 + threadIdx.y * 4 + j;
        dst[(long)e * Ld + l0 + threadIdx.x] = tile[threadIdx.x][threadIdx.y * 4 + j];
    }
}

// ---------------------------------------------------------------------------
// 6) ctx: per (h,n), CTX[l,e] = sum_m P[l,m] * V[m,e]  (bf16 out)
//    M=Ld, N=Ed, K=Ld
// ---------------------------------------------------------------------------
__global__ __launch_bounds__(256) void k_gemm_ctx(const __bf16* __restrict__ P,
                                                  const __bf16* __restrict__ VT,
                                                  __bf16* __restrict__ CTX) {
    __shared__ alignas(16) __bf16 sA[2][256 * 32];
    __shared__ alignas(16) __bf16 sB[2][64 * 32];
    const int b = blockIdx.z, h = b >> 2, n = b & 3;
    const int wave = threadIdx.x >> 5;
    const int lane = threadIdx.x & 31;
    const int row0 = blockIdx.y * 256;   // l
    const int col0 = blockIdx.x * 64;    // e

    const __bf16* A  = P  + (long)b * Ld * Ld;
    const __bf16* Bt = VT + (long)b * Ed * Ld;       // rows = e, K contiguous
    __bf16* C = CTX + (long)h * Td * Ed + (long)n * Ed;

    f32x8 acc[2][4] = {};
    block_gemm_lds(A, Ld, Bt, Ld, row0, col0, Ld, sA, sB, acc);

    const int cn = lane & 15, rh = (lane >> 4) * 8;
#pragma unroll
    for (int rr = 0; rr < 2; ++rr) {
        const int row = row0 + wave * 32 + rr * 16;
#pragma unroll
        for (int j = 0; j < 4; ++j)
#pragma unroll
            for (int i = 0; i < 8; ++i)
                C[(long)(row + rh + i) * (Nd * Ed) + col0 + 16 * j + cn] = f2bf(acc[rr][j][i]);
    }
}

// ---------------------------------------------------------------------------
// 7) Output proj + head sum: out[t,f] = sum_h ( CTX[h] * Wo[h]^T )[t,f] + sum_h bo[h,f]
//    M=Td, N=Ed, K = Hd * Ed (loop over heads accumulating into the same tile)
// ---------------------------------------------------------------------------
__global__ __launch_bounds__(256) void k_gemm_out(const __bf16* __restrict__ CTX,
                                                  const __bf16* __restrict__ WO,
                                                  const float* __restrict__ bo,
                                                  float* __restrict__ OUT) {
    __shared__ alignas(16) __bf16 sA[2][256 * 32];
    __shared__ alignas(16) __bf16 sB[2][64 * 32];
    const int wave = threadIdx.x >> 5;
    const int lane = threadIdx.x & 31;
    const int row0 = blockIdx.y * 256;   // t
    const int col0 = blockIdx.x * 64;    // f

    f32x8 acc[2][4] = {};
    for (int h = 0; h < Hd; ++h) {
        const __bf16* A  = CTX + (long)h * Td * Ed;
        const __bf16* Bt = WO  + (long)h * Ed * Ed;
        block_gemm_lds(A, Ed, Bt, Ed, row0, col0, Ed, sA, sB, acc);
    }

    const int cn = lane & 15, rh = (lane >> 4) * 8;
#pragma unroll
    for (int rr = 0; rr < 2; ++rr) {
        const int row = row0 + wave * 32 + rr * 16;
#pragma unroll
        for (int j = 0; j < 4; ++j) {
            const int col = col0 + 16 * j + cn;
            float bias = 0.f;
#pragma unroll
            for (int h = 0; h < Hd; ++h) bias += bo[h * Ed + col];
#pragma unroll
            for (int i = 0; i < 8; ++i)
                OUT[(long)(row + rh + i) * Ed + col] = acc[rr][j][i] + bias;
        }
    }
}

// ---------------------------------------------------------------------------
extern "C" void kernel_launch(void* const* d_in, const int* in_sizes, int n_in,
                              void* d_out, int out_size, void* d_ws, size_t ws_size,
                              hipStream_t stream) {
    const float* x    = (const float*)d_in[0];
    const float* Wqkv = (const float*)d_in[1];
    const float* bqkv = (const float*)d_in[2];
    const float* Wo   = (const float*)d_in[3];
    const float* bo   = (const float*)d_in[4];
    float* out = (float*)d_out;

    char* ws = (char*)d_ws;
    __bf16* xb   = (__bf16*)(ws + OFF_XB);
    __bf16* wqb  = (__bf16*)(ws + OFF_WQKVB);
    __bf16* wob  = (__bf16*)(ws + OFF_WOB);
    __bf16* qkv  = (__bf16*)(ws + OFF_QKV);
    float*  sc   = (float*)(ws + OFF_SC);
    __bf16* attn = (__bf16*)(ws + OFF_AT);
    __bf16* vt   = (__bf16*)(ws + OFF_VT);
    __bf16* ctx  = (__bf16*)(ws + OFF_CTX);

    // 1) f32 -> bf16
    {
        long n;
        n = (long)Td * Ed;      k_cvt_bf16<<<(int)(n / 1024), 256, 0, stream>>>(x, xb, n);
        n = (long)Hd * F3 * Ed; k_cvt_bf16<<<(int)(n / 1024), 256, 0, stream>>>(Wqkv, wqb, n);
        n = (long)Hd * Ed * Ed; k_cvt_bf16<<<(int)(n / 1024), 256, 0, stream>>>(Wo, wob, n);
    }
    // 2) QKV projection
    k_gemm_qkv<<<dim3(F3 / 64, Td / 256, Hd), 256, 0, stream>>>(xb, wqb, bqkv, qkv);
    // 3) scores
    k_gemm_scores<<<dim3(Ld / 64, Ld / 256, BHN), 256, 0, stream>>>(qkv, sc);
    // 4) softmax
    k_softmax<<<BHN * Ld, 256, 0, stream>>>(sc, attn);
    // 5) V transpose
    k_transpose_v<<<dim3(Ld / 32, Ed / 32, BHN), dim3(32, 8), 0, stream>>>(qkv, vt);
    // 6) ctx
    k_gemm_ctx<<<dim3(Ed / 64, Ld / 256, BHN), 256, 0, stream>>>(attn, vt, ctx);
    // 7) output projection + head sum
    k_gemm_out<<<dim3(Ed / 64, Td / 256, 1), 256, 0, stream>>>(ctx, wob, bo, out);
}
